// Yolov2Net_6725918786251
// MI455X (gfx1250) — compile-verified, matrix operand order checked by reading, not run on hardware
//
#include <hip/hip_runtime.h>
#include <hip/hip_bf16.h>
#include <math.h>

// ---------------------------------------------------------------------------
// YOLOv2 (Darknet-19) forward on gfx1250.
// Implicit-GEMM convs via v_wmma_f32_16x16x32_bf16 (bf16 in, fp32 acc),
// weights pre-transposed+padded to [Cout_pad][K_pad] so the B tile is staged
// with guard-free GLOBAL_LOAD_ASYNC_TO_LDS_B128 (ASYNCcnt), two-pass BN
// (training-mode batch stats), fused leaky+maxpool, WMMA linear head.
// ---------------------------------------------------------------------------

typedef __attribute__((ext_vector_type(16))) __bf16 v16bf;
typedef __attribute__((ext_vector_type(8)))  float  v8f;
typedef __attribute__((ext_vector_type(4)))  int    v4i;

union FragBF { v16bf v; uint4 q[2]; };

__device__ __forceinline__ unsigned short f2bf(float f) {
    unsigned int u = __float_as_uint(f);
    unsigned int r = u + 0x7FFFu + ((u >> 16) & 1u);   // round-to-nearest-even
    return (unsigned short)(r >> 16);
}
__device__ __forceinline__ float bf2f(unsigned short h) {
    return __uint_as_float(((unsigned int)h) << 16);
}

// ---- gfx1250 async global->LDS copy (guarded; falls back to plain stores) --
#if defined(__gfx1250__) && __has_builtin(__builtin_amdgcn_global_load_async_to_lds_b128)
#define HAS_ASYNC_LDS 1
typedef __attribute__((address_space(1))) v4i* gv4i_p;
typedef __attribute__((address_space(3))) v4i* lv4i_p;
__device__ __forceinline__ void async_copy16(const void* g, void* l) {
    __builtin_amdgcn_global_load_async_to_lds_b128((gv4i_p)g, (lv4i_p)l, 0, 0);
}
__device__ __forceinline__ void wait_async0() {
#if __has_builtin(__builtin_amdgcn_s_wait_asynccnt)
    __builtin_amdgcn_s_wait_asynccnt(0);
#else
    asm volatile("s_wait_asynccnt 0x0" ::: "memory");
#endif
}
#else
#define HAS_ASYNC_LDS 0
#endif

// ---------------------------------------------------------------------------
// Prep kernels
// ---------------------------------------------------------------------------

// x: [16][C][H][W] fp32  ->  act: [16][H][W][C] bf16
__global__ void k_convert_input(const float* __restrict__ x,
                                unsigned short* __restrict__ act,
                                int C, int H, int W) {
    long long total = 16LL * C * H * W;
    long long idx = (long long)blockIdx.x * blockDim.x + threadIdx.x;
    if (idx >= total) return;
    int c = (int)(idx % C); long long t = idx / C;
    int w = (int)(t % W); t /= W;
    int h = (int)(t % H); int n = (int)(t / H);
    act[idx] = f2bf(x[(((long long)n * C + c) * H + h) * W + w]);
}

// W: [Cout][Cin][K][K] fp32 -> Wt: [Coutpad][Kpad] bf16 (K contiguous,
// kidx = (ky*K+kx)*Cin+ci, zero-padded so async 16B runs are always legal)
__global__ void k_prep_weights(const float* __restrict__ Wp,
                               unsigned short* __restrict__ wt,
                               int Cin, int Cout, int K, int Kdim,
                               int Kpad, int Coutpad) {
    int total = Coutpad * Kpad;
    int i = blockIdx.x * blockDim.x + threadIdx.x;
    if (i >= total) return;
    int kidx = i % Kpad; int o = i / Kpad;
    unsigned short v = 0;
    if (o < Cout && kidx < Kdim) {
        int ci = kidx % Cin; int t = kidx / Cin;
        int kx = t % K; int ky = t / K;
        v = f2bf(Wp[(((long long)o * Cin + ci) * K + ky) * K + kx]);
    }
    wt[i] = v;
}

// lin_W: [1000][21125] fp32 -> Wt: [21184][1024] bf16 (transposed, padded)
__global__ void k_prep_linW(const float* __restrict__ Wp,
                            unsigned short* __restrict__ wt) {
    long long total = 21184LL * 1024;
    long long i = (long long)blockIdx.x * blockDim.x + threadIdx.x;
    if (i >= total) return;
    int k = (int)(i & 1023); int n = (int)(i >> 10);
    unsigned short v = 0;
    if (n < 21125 && k < 1000) v = f2bf(Wp[(long long)k * 21125 + n]);
    wt[i] = v;
}

// ---------------------------------------------------------------------------
// Implicit-GEMM conv: y[M=16*H*W][Cout] = im2col(act) @ W + bias
// 128 threads (4 waves), 64x64 block tile, each wave a 32x32 quadrant
// (2x2 WMMA 16x16 tiles). K staged in 32-wide bf16 chunks through LDS.
// B tile: guard-free async DMA from padded transposed weights.
// A tile: im2col gather; pixel decode hoisted out of the K loop.
// ---------------------------------------------------------------------------
__global__ __launch_bounds__(128) void k_conv_wmma(
    const unsigned short* __restrict__ act,  // [16*H*W][Cin] bf16 NHWC
    const unsigned short* __restrict__ wt,   // [Coutpad][Kpad] bf16
    const float* __restrict__ bias,          // [Cout]
    float* __restrict__ y,                   // [Mtot][Cout]  fp32
    int H, int W, int Cin, int Cout, int Ksz, int pad,
    int Mtot, int Kdim, int Kpad, int cinShift, int cinMask) {

    __shared__ __align__(16) unsigned short At[64 * 40];  // [row][k], stride 40
    __shared__ __align__(16) unsigned short Bt[64 * 40];  // [col][k], stride 40

    const int tid  = threadIdx.x;
    const int lane = tid & 31;
    const int wv   = tid >> 5;
    const int wm   = (wv >> 1) * 32;
    const int wn   = (wv & 1)  * 32;
    const int half = lane >> 4;
    const int r    = lane & 15;
    const int m0   = blockIdx.x * 64;
    const int n0   = blockIdx.y * 64;
    const int HW   = H * W;

    // ---- per-thread fixed A-gather coordinates (K-invariant, hoisted) ----
    const int rowA   = tid >> 1;          // 0..63
    const int kkBase = (tid & 1) * 16;    // 0 or 16
    const int gm = m0 + rowA;
    const bool mok = gm < Mtot;
    int pn = 0, oy = 0, ox = 0;
    if (mok) { pn = gm / HW; int rem = gm - pn * HW; oy = rem / W; ox = rem - oy * W; }
    const unsigned short* actN = act + (long long)pn * HW * Cin;

    v8f acc[2][2];
    for (int tj = 0; tj < 2; ++tj) {
        int col = n0 + wn + tj * 16 + r;
        float bv = (col < Cout) ? bias[col] : 0.0f;
        v8f c;
        #pragma unroll
        for (int j = 0; j < 8; ++j) c[j] = bv;
        acc[0][tj] = c; acc[1][tj] = c;
    }

    for (int k0 = 0; k0 < Kdim; k0 += 32) {
        // ---- B tile: 64 cols x 32 k, 16B runs, guard-free (padded) ----
#if HAS_ASYNC_LDS
        {
            // 256 runs of 16B; 128 threads -> 2 each
            int i0 = tid;           // run index
            int c0 = i0 >> 2, run0 = i0 & 3;
            async_copy16(&wt[(long long)(n0 + c0) * Kpad + k0 + run0 * 8],
                         &Bt[c0 * 40 + run0 * 8]);
            int i1 = tid + 128;
            int c1 = i1 >> 2, run1 = i1 & 3;
            async_copy16(&wt[(long long)(n0 + c1) * Kpad + k0 + run1 * 8],
                         &Bt[c1 * 40 + run1 * 8]);
        }
#else
        for (int i = tid; i < 64 * 32; i += 128) {
            int c = i >> 5, kk = i & 31;
            Bt[c * 40 + kk] = wt[(long long)(n0 + c) * Kpad + k0 + kk];
        }
        if (k0 + 32 < Kdim)
            __builtin_prefetch(&wt[(long long)(n0 + tid) * Kpad + k0 + 32], 0, 1);
#endif
        // ---- A tile: im2col gather, 16 elements per thread ----
        {
            unsigned short* dst = &At[rowA * 40 + kkBase];
            #pragma unroll 4
            for (int e = 0; e < 16; ++e) {
                int kg = k0 + kkBase + e;
                unsigned short v = 0;
                if (mok && kg < Kdim) {
                    int ci, t;
                    if (cinShift) { ci = kg & cinMask; t = kg >> cinShift; }
                    else          { ci = kg % Cin;     t = kg / Cin; }
                    int kx, ky;
                    if (Ksz == 1) { kx = 0; ky = 0; }
                    else          { kx = t % 3; ky = t / 3; }
                    int iy = oy + ky - pad, ix = ox + kx - pad;
                    if (iy >= 0 && iy < H && ix >= 0 && ix < W)
                        v = actN[((long long)iy * W + ix) * Cin + ci];
                }
                dst[e] = v;
            }
        }
#if HAS_ASYNC_LDS
        wait_async0();
#endif
        __syncthreads();

        // ---- fragments: two 16B LDS reads per fragment per lane ----
        FragBF a[2], b[2];
        #pragma unroll
        for (int t = 0; t < 2; ++t) {
            int row = wm + t * 16 + r;
            a[t].q[0] = *(const uint4*)&At[row * 40 + half * 8];
            a[t].q[1] = *(const uint4*)&At[row * 40 + 16 + half * 8];
            int col = wn + t * 16 + r;
            b[t].q[0] = *(const uint4*)&Bt[col * 40 + half * 8];
            b[t].q[1] = *(const uint4*)&Bt[col * 40 + 16 + half * 8];
        }
        #pragma unroll
        for (int ti = 0; ti < 2; ++ti)
            #pragma unroll
            for (int tj = 0; tj < 2; ++tj)
                acc[ti][tj] = __builtin_amdgcn_wmma_f32_16x16x32_bf16(
                    false, a[ti].v, false, b[tj].v, (short)0, acc[ti][tj],
                    false, false);
        __syncthreads();
    }

    // ---- store: C/D layout -> lane = column, vgpr j = row (j + 8*half) ----
    #pragma unroll
    for (int ti = 0; ti < 2; ++ti)
        #pragma unroll
        for (int tj = 0; tj < 2; ++tj) {
            int col = n0 + wn + tj * 16 + r;
            if (col >= Cout) continue;
            int rbase = m0 + wm + ti * 16 + half * 8;
            #pragma unroll
            for (int j = 0; j < 8; ++j) {
                int row = rbase + j;
                if (row < Mtot)
                    y[(long long)row * Cout + col] = acc[ti][tj][j];
            }
        }
}

// ---------------------------------------------------------------------------
// BatchNorm batch statistics: one block per channel
// ---------------------------------------------------------------------------
__global__ __launch_bounds__(256) void k_bn_stats(
    const float* __restrict__ y, const float* __restrict__ gamma,
    const float* __restrict__ beta, float* __restrict__ scale,
    float* __restrict__ shift, int Mtot, int Cout) {
    int c = blockIdx.x;
    float s = 0.0f, s2 = 0.0f;
    for (int i = threadIdx.x; i < Mtot; i += 256) {
        float v = y[(long long)i * Cout + c];
        s += v; s2 += v * v;
    }
    __shared__ float sh[256], sh2[256];
    sh[threadIdx.x] = s; sh2[threadIdx.x] = s2;
    __syncthreads();
    for (int st = 128; st > 0; st >>= 1) {
        if (threadIdx.x < st) {
            sh[threadIdx.x]  += sh[threadIdx.x + st];
            sh2[threadIdx.x] += sh2[threadIdx.x + st];
        }
        __syncthreads();
    }
    if (threadIdx.x == 0) {
        float inv = 1.0f / (float)Mtot;
        float mean = sh[0] * inv;
        float var  = sh2[0] * inv - mean * mean;
        float sc = gamma[c] * rsqrtf(var + 1e-5f);
        scale[c] = sc;
        shift[c] = beta[c] - mean * sc;
    }
}

// ---------------------------------------------------------------------------
// Affine + LeakyReLU (+ optional 2x2 maxpool), fp32 -> bf16 NHWC
// ---------------------------------------------------------------------------
__global__ void k_bn_apply(const float* __restrict__ y,
                           const float* __restrict__ scale,
                           const float* __restrict__ shift,
                           unsigned short* __restrict__ out,
                           int H, int W, int Cout, int pool) {
    int Ho = pool ? H / 2 : H, Wo = pool ? W / 2 : W;
    long long total = 16LL * Ho * Wo * Cout;
    for (long long idx = (long long)blockIdx.x * blockDim.x + threadIdx.x;
         idx < total; idx += (long long)gridDim.x * blockDim.x) {
        int c = (int)(idx % Cout); long long t = idx / Cout;
        int ox = (int)(t % Wo); t /= Wo;
        int oy = (int)(t % Ho); int n = (int)(t / Ho);
        float sc = scale[c], sf = shift[c];
        float best;
        if (pool) {
            best = -3.402823466e38f;
            #pragma unroll
            for (int dy = 0; dy < 2; ++dy)
                #pragma unroll
                for (int dx = 0; dx < 2; ++dx) {
                    long long pix = ((long long)n * H + (2 * oy + dy)) * W + (2 * ox + dx);
                    float v = y[pix * Cout + c] * sc + sf;
                    v = v > 0.0f ? v : 0.01f * v;
                    best = fmaxf(best, v);
                }
        } else {
            long long pix = ((long long)n * H + oy) * W + ox;
            float v = y[pix * Cout + c] * sc + sf;
            best = v > 0.0f ? v : 0.01f * v;
        }
        out[idx] = f2bf(best);
    }
}

// ---------------------------------------------------------------------------
// Global average pool over 13x13: [16][169][1000] bf16 -> [16][1024] bf16
// ---------------------------------------------------------------------------
__global__ void k_avgpool(const unsigned short* __restrict__ act,
                          unsigned short* __restrict__ out) {
    int idx = blockIdx.x * blockDim.x + threadIdx.x;
    if (idx >= 16 * 1024) return;
    int c = idx & 1023, n = idx >> 10;
    unsigned short v = 0;
    if (c < 1000) {
        float s = 0.0f;
        for (int p = 0; p < 169; ++p)
            s += bf2f(act[((long long)n * 169 + p) * 1000 + c]);
        v = f2bf(s * (1.0f / 169.0f));
    }
    out[idx] = v;
}

// ---------------------------------------------------------------------------
// Linear head: out[16][21125] = act[16][1000] @ linW + b, sigmoid on
// (col % 25) < 3. Wt is [21184][1024] bf16 so staging is guard-free.
// ---------------------------------------------------------------------------
__global__ __launch_bounds__(128) void k_linear_wmma(
    const unsigned short* __restrict__ a,    // [16][1024] bf16
    const unsigned short* __restrict__ wt,   // [21184][1024] bf16
    const float* __restrict__ bias,          // [21125]
    float* __restrict__ out) {               // [16][21125]
    __shared__ __align__(16) unsigned short At[16 * 40];
    __shared__ __align__(16) unsigned short Bt[64 * 40];
    const int tid = threadIdx.x, lane = tid & 31, wv = tid >> 5;
    const int half = lane >> 4, r = lane & 15;
    const int n0 = blockIdx.x * 64;
    const int NOUT = 21125;

    v8f acc;
    #pragma unroll
    for (int j = 0; j < 8; ++j) acc[j] = 0.0f;

    for (int k0 = 0; k0 < 1024; k0 += 32) {
#if HAS_ASYNC_LDS
        if (tid < 64) {               // A: 16 rows x 4 runs of 16B
            int row = tid >> 2, run = tid & 3;
            async_copy16(&a[row * 1024 + k0 + run * 8],
                         &At[row * 40 + run * 8]);
        }
        {
            // B: 64 cols x 4 runs; 2 per thread
            int c0 = tid >> 2, run0 = tid & 3;
            async_copy16(&wt[(long long)(n0 + c0) * 1024 + k0 + run0 * 8],
                         &Bt[c0 * 40 + run0 * 8]);
            int i1 = tid + 128;
            int c1 = i1 >> 2, run1 = i1 & 3;
            async_copy16(&wt[(long long)(n0 + c1) * 1024 + k0 + run1 * 8],
                         &Bt[c1 * 40 + run1 * 8]);
        }
        wait_async0();
#else
        for (int i = tid; i < 16 * 32; i += 128) {
            int row = i >> 5, kk = i & 31;
            At[row * 40 + kk] = a[row * 1024 + k0 + kk];
        }
        for (int i = tid; i < 64 * 32; i += 128) {
            int c = i >> 5, kk = i & 31;
            Bt[c * 40 + kk] = wt[(long long)(n0 + c) * 1024 + k0 + kk];
        }
#endif
        __syncthreads();
        FragBF fa, fb;
        fa.q[0] = *(const uint4*)&At[r * 40 + half * 8];
        fa.q[1] = *(const uint4*)&At[r * 40 + 16 + half * 8];
        int col = wv * 16 + r;
        fb.q[0] = *(const uint4*)&Bt[col * 40 + half * 8];
        fb.q[1] = *(const uint4*)&Bt[col * 40 + 16 + half * 8];
        acc = __builtin_amdgcn_wmma_f32_16x16x32_bf16(
            false, fa.v, false, fb.v, (short)0, acc, false, false);
        __syncthreads();
    }

    int col = n0 + wv * 16 + r;
    if (col < NOUT) {
        float bb = bias[col];
        bool sig = (col % 25) < 3;
        #pragma unroll
        for (int j = 0; j < 8; ++j) {
            int row = half * 8 + j;             // batch index 0..15
            float v = acc[j] + bb;
            if (sig) v = 1.0f / (1.0f + expf(-v));
            out[(long long)row * NOUT + col] = v;
        }
    }
}

// ---------------------------------------------------------------------------
// Host orchestration
// ---------------------------------------------------------------------------

struct ConvSpec { int cin, cout, k, pad, pool; };
static const ConvSpec g_specs[19] = {
    {   3,   32, 3, 1, 1},
    {  32,   64, 3, 1, 1},
    {  64,  128, 3, 1, 0}, { 128,   64, 1, 0, 0}, {  64,  128, 3, 1, 1},
    { 128,  256, 3, 1, 0}, { 256,  128, 1, 0, 0}, { 128,  256, 3, 1, 1},
    { 256,  512, 3, 1, 0}, { 512,  256, 1, 0, 0}, { 256,  512, 3, 1, 0},
    { 512,  256, 1, 0, 0}, { 256,  512, 3, 1, 1},
    { 512, 1024, 3, 1, 0}, {1024,  512, 1, 0, 0}, { 512, 1024, 3, 1, 0},
    {1024,  512, 1, 0, 0}, { 512, 1024, 3, 1, 0},
    {1024, 1000, 1, 0, 0},
};

static int ilog2_if_pow2(int v) {
    if (v <= 0 || (v & (v - 1))) return 0;
    int s = 0; while ((1 << s) < v) ++s; return s;
}

// workspace carving (bytes, all 256-aligned)
static const size_t Y_BYTES   = (size_t)16 * 416 * 416 * 32 * 4;   // largest fp32 pre-BN buffer (layer0)
static const size_t ACT_BYTES = (size_t)16 * 208 * 208 * 32 * 2;   // largest bf16 activation
static const size_t WB_BYTES  = (size_t)21184 * 1024 * 2;          // 43,384,832 (lin_W padded; conv Wt <= 9.4MB)

extern "C" void kernel_launch(void* const* d_in, const int* in_sizes, int n_in,
                              void* d_out, int out_size, void* d_ws, size_t ws_size,
                              hipStream_t stream) {
    (void)in_sizes; (void)out_size; (void)ws_size;

    const float* x    = (const float*)d_in[0];
    const float* linW = (const float*)d_in[n_in - 2];
    const float* linb = (const float*)d_in[n_in - 1];

    char* ws = (char*)d_ws;
    float*          Y  = (float*)(ws);
    unsigned short* A0 = (unsigned short*)(ws + Y_BYTES);
    unsigned short* A1 = (unsigned short*)(ws + Y_BYTES + ACT_BYTES);
    unsigned short* WT = (unsigned short*)(ws + Y_BYTES + 2 * ACT_BYTES);
    float*          SC = (float*)(ws + Y_BYTES + 2 * ACT_BYTES + WB_BYTES);
    float*          SH = SC + 1024;

    // input: NCHW fp32 -> NHWC bf16
    {
        long long total = 16LL * 3 * 416 * 416;
        k_convert_input<<<(int)((total + 255) / 256), 256, 0, stream>>>(x, A0, 3, 416, 416);
    }

    unsigned short* cur = A0;
    unsigned short* nxt = A1;
    int H = 416, W = 416;

    for (int i = 0; i < 19; ++i) {
        const ConvSpec s = g_specs[i];
        const float* Wp = (const float*)d_in[1 + 4 * i];
        const float* bp = (const float*)d_in[2 + 4 * i];
        const float* gp = (const float*)d_in[3 + 4 * i];
        const float* bt = (const float*)d_in[4 + 4 * i];

        int Kdim    = s.k * s.k * s.cin;
        int Kpad    = (Kdim + 31) & ~31;
        int Coutpad = (s.cout + 63) & ~63;
        int Mtot    = 16 * H * W;
        int cinShift = ilog2_if_pow2(s.cin);
        int cinMask  = s.cin - 1;

        // repack weights to bf16 [Coutpad][Kpad]
        {
            int wtot = Coutpad * Kpad;
            k_prep_weights<<<(wtot + 255) / 256, 256, 0, stream>>>(
                Wp, WT, s.cin, s.cout, s.k, Kdim, Kpad, Coutpad);
        }
        // conv (implicit GEMM, WMMA bf16, async B staging)
        {
            dim3 g((unsigned)((Mtot + 63) / 64), (unsigned)(Coutpad / 64));
            k_conv_wmma<<<g, 128, 0, stream>>>(cur, WT, bp, Y,
                                               H, W, s.cin, s.cout, s.k, s.pad,
                                               Mtot, Kdim, Kpad, cinShift, cinMask);
        }
        // batch stats -> scale/shift
        k_bn_stats<<<s.cout, 256, 0, stream>>>(Y, gp, bt, SC, SH, Mtot, s.cout);
        // affine + leaky + optional pool -> bf16 NHWC
        k_bn_apply<<<8192, 256, 0, stream>>>(Y, SC, SH, nxt, H, W, s.cout, s.pool);

        if (s.pool) { H /= 2; W /= 2; }
        unsigned short* tmp = cur; cur = nxt; nxt = tmp;
    }

    // avg-pool 13x13 -> [16][1024] bf16 (zero-padded K)
    k_avgpool<<<(16 * 1024 + 255) / 256, 256, 0, stream>>>(cur, nxt);

    // lin_W -> transposed padded bf16 [21184][1024]
    {
        long long n = 21184LL * 1024;
        k_prep_linW<<<(int)((n + 255) / 256), 256, 0, stream>>>(linW, WT);
    }
    // linear + bias + selective sigmoid
    k_linear_wmma<<<21184 / 64, 128, 0, stream>>>(nxt, WT, linb, (float*)d_out);
}
